// FFTMLP_41669772706066
// MI455X (gfx1250) — matrix operand order repaired and checked by reading, not compiled, not run on hardware
//
#include <hip/hip_runtime.h>

// ---- problem constants (from reference) ----
#define NN      32768
#define TT      2048
#define HID     128
#define NF      1025      // TT/2 + 1
#define FAN1    2050      // 2*NF

typedef __bf16 bf16_t;
typedef __attribute__((ext_vector_type(8)))  __bf16 v8bf;
typedef __attribute__((ext_vector_type(16))) __bf16 v16bf;
typedef __attribute__((ext_vector_type(8)))  float  v8f;

// -------------------------------------------------------------------------
// Kernel 1: v[h] = mean_t W2[t,h];  c = mean(b2)
// -------------------------------------------------------------------------
__global__ __launch_bounds__(HID) void fftmlp_prep_v(
    const float* __restrict__ W2, const float* __restrict__ b2,
    float* __restrict__ v, float* __restrict__ cptr)
{
    const int h = threadIdx.x;
    float s = 0.f;
    for (int t = 0; t < TT; ++t) s += W2[t * HID + h];   // coalesced across h
    v[h] = s * (1.f / (float)TT);

    __shared__ float red[HID];
    float cb = 0.f;
    for (int i = h; i < TT; i += HID) cb += b2[i];
    red[h] = cb;
    __syncthreads();
    for (int off = HID / 2; off > 0; off >>= 1) {
        if (h < off) red[h] += red[h + off];
        __syncthreads();
    }
    if (h == 0) *cptr = red[0] * (1.f / (float)TT);
}

// -------------------------------------------------------------------------
// Kernel 2: A[h][t] = sum_k W1[h,k]*cos(2*pi*k*t/T) - W1[h,NF+k]*sin(...)
// One block per t-column; sincos staged in LDS, shared by all 128 h threads.
// Output stored bf16, row-major [HID][TT].
// -------------------------------------------------------------------------
__global__ __launch_bounds__(HID) void fftmlp_build_A(
    const float* __restrict__ W1, bf16_t* __restrict__ A)
{
    const int t = blockIdx.x;
    const int h = threadIdx.x;
    __shared__ float cs[HID];
    __shared__ float sn[HID];

    const float* w1r = W1 + (size_t)h * FAN1;
    const float w = 6.283185307179586f / (float)TT;

    float acc = 0.f;
    for (int kb = 0; kb < NF; kb += HID) {
        const int valid = (NF - kb) < HID ? (NF - kb) : HID;
        __syncthreads();                       // protect previous chunk reads
        if (h < valid) {
            const int k  = kb + h;
            const int ph = (k * t) & (TT - 1); // exact mod, T is pow2
            float sv, cv;
            __sincosf((float)ph * w, &sv, &cv);
            cs[h] = cv; sn[h] = sv;
        }
        __syncthreads();
        for (int i = 0; i < valid; ++i) {
            const int k = kb + i;
            acc = fmaf(w1r[k],       cs[i], acc);
            acc = fmaf(-w1r[NF + k], sn[i], acc);
        }
    }
    A[(size_t)h * TT + t] = (bf16_t)acc;
}

// -------------------------------------------------------------------------
// Kernel 3: fused  diff -> GEMM(WMMA bf16) -> bias/relu -> dot(v) -> +c
// Block = 256 threads = 8 waves; block owns 128 samples, wave owns 16.
// LDS row stride 40 halves (80 B) -> conflict-free, 16B-aligned b128 loads.
// -------------------------------------------------------------------------
#define LSTR 40

__global__ __launch_bounds__(256) void fftmlp_main(
    const float* __restrict__ x1, const float* __restrict__ x2,
    const bf16_t* __restrict__ A, const float* __restrict__ b1,
    const float* __restrict__ vvec, const float* __restrict__ cptr,
    float* __restrict__ out)
{
    __shared__ __align__(16) bf16_t sD[128 * LSTR];  // diff chunk  [128 samp][32 t]
    __shared__ __align__(16) bf16_t sB[HID * LSTR];  // A     chunk [128 h   ][32 t]

    const int tid  = threadIdx.x;
    const int lane = tid & 31;
    const int wv   = tid >> 5;      // 0..7
    const int row  = lane & 15;     // M / N index within fragment
    const int hi   = lane >> 4;     // lane-half select (K group)
    const int S0   = blockIdx.x * 128;

    // cooperative-loader mapping: 2 threads per row, 16 floats each
    const int lrow = tid >> 1;
    const int lcol = (tid & 1) * 16;
    const float*  px1 = x1 + (size_t)(S0 + lrow) * TT + lcol;
    const float*  px2 = x2 + (size_t)(S0 + lrow) * TT + lcol;
    const bf16_t* pA  = A  + (size_t)lrow * TT + lcol;
    bf16_t* dstD = sD + lrow * LSTR + lcol;
    bf16_t* dstB = sB + lrow * LSTR + lcol;

    v8f acc[8];
    const v8f vz = {0.f,0.f,0.f,0.f,0.f,0.f,0.f,0.f};
#pragma unroll
    for (int j = 0; j < 8; ++j) acc[j] = vz;

    for (int t0 = 0; t0 < TT; t0 += 32) {
        // ---- stage diff chunk (fp32 loads, bf16 LDS) ----
        float4 a0 = *(const float4*)(px1 + t0 + 0);
        float4 a1 = *(const float4*)(px1 + t0 + 4);
        float4 a2 = *(const float4*)(px1 + t0 + 8);
        float4 a3 = *(const float4*)(px1 + t0 + 12);
        float4 c0 = *(const float4*)(px2 + t0 + 0);
        float4 c1 = *(const float4*)(px2 + t0 + 4);
        float4 c2 = *(const float4*)(px2 + t0 + 8);
        float4 c3 = *(const float4*)(px2 + t0 + 12);
        float df[16] = { a0.x-c0.x, a0.y-c0.y, a0.z-c0.z, a0.w-c0.w,
                         a1.x-c1.x, a1.y-c1.y, a1.z-c1.z, a1.w-c1.w,
                         a2.x-c2.x, a2.y-c2.y, a2.z-c2.z, a2.w-c2.w,
                         a3.x-c3.x, a3.y-c3.y, a3.z-c3.z, a3.w-c3.w };
        v8bf d0, d1;
#pragma unroll
        for (int i = 0; i < 8; ++i) { d0[i] = (bf16_t)df[i]; d1[i] = (bf16_t)df[8 + i]; }
        *(v8bf*)(dstD)     = d0;
        *(v8bf*)(dstD + 8) = d1;

        // ---- stage shared A chunk ----
        *(v8bf*)(dstB)     = *(const v8bf*)(pA + t0);
        *(v8bf*)(dstB + 8) = *(const v8bf*)(pA + t0 + 8);

        if (t0 + 32 < TT) {
            __builtin_prefetch(px1 + t0 + 32, 0, 1);
            __builtin_prefetch(px2 + t0 + 32, 0, 1);
        }
        __syncthreads();

        // ---- A fragment: lane L -> row M=L%16; K = hi*8 + {0..7, 16..23} ----
        const bf16_t* ar = sD + (wv * 16 + row) * LSTR + hi * 8;
        v8bf fa0 = *(const v8bf*)(ar);
        v8bf fa1 = *(const v8bf*)(ar + 16);
        v16bf fa = __builtin_shufflevector(fa0, fa1, 0,1,2,3,4,5,6,7,8,9,10,11,12,13,14,15);

#pragma unroll
        for (int j = 0; j < 8; ++j) {
            // B fragment: lane L -> col N=L%16 (h), K = hi*8 + {0..7, 16..23}
            const bf16_t* br = sB + (j * 16 + row) * LSTR + hi * 8;
            v8bf fb0 = *(const v8bf*)(br);
            v8bf fb1 = *(const v8bf*)(br + 16);
            v16bf fb = __builtin_shufflevector(fb0, fb1, 0,1,2,3,4,5,6,7,8,9,10,11,12,13,14,15);
            acc[j] = __builtin_amdgcn_wmma_f32_16x16x32_bf16(
                         false, fa, false, fb, (short)0, acc[j], false, false);
        }
        __syncthreads();
    }

    // ---- epilogue: +b1, relu, *v[h], reduce over h ----
    const float cbar = *cptr;
    float sum[8];
#pragma unroll
    for (int r = 0; r < 8; ++r) sum[r] = 0.f;
#pragma unroll
    for (int j = 0; j < 8; ++j) {
        const int h  = j * 16 + row;
        const float bb = b1[h];
        const float vv = vvec[h];
#pragma unroll
        for (int r = 0; r < 8; ++r) {
            float x = acc[j][r] + bb;
            x = x > 0.f ? x : 0.f;
            sum[r] = fmaf(x, vv, sum[r]);
        }
    }
#pragma unroll
    for (int r = 0; r < 8; ++r) {
        float s = sum[r];
        s += __shfl_xor(s, 1, 16);
        s += __shfl_xor(s, 2, 16);
        s += __shfl_xor(s, 4, 16);
        s += __shfl_xor(s, 8, 16);
        sum[r] = s;
    }
    if (row == 0) {                       // lanes 0 and 16
        const int base = S0 + wv * 16 + hi * 8;   // C layout: lanes16-31 -> M=r+8
#pragma unroll
        for (int r = 0; r < 8; ++r) out[base + r] = sum[r] + cbar;
    }
}

// -------------------------------------------------------------------------
extern "C" void kernel_launch(void* const* d_in, const int* in_sizes, int n_in,
                              void* d_out, int out_size, void* d_ws, size_t ws_size,
                              hipStream_t stream)
{
    const float* x1 = (const float*)d_in[0];
    const float* x2 = (const float*)d_in[1];
    const float* W1 = (const float*)d_in[2];
    const float* b1 = (const float*)d_in[3];
    const float* W2 = (const float*)d_in[4];
    const float* b2 = (const float*)d_in[5];
    float* out = (float*)d_out;

    char* ws = (char*)d_ws;
    bf16_t* A    = (bf16_t*)ws;                                  // 128*2048*2 = 512 KB
    float*  v    = (float*)(ws + (size_t)HID * TT * sizeof(bf16_t));
    float*  cpt  = v + HID;

    fftmlp_prep_v <<<1,    HID, 0, stream>>>(W2, b2, v, cpt);
    fftmlp_build_A<<<TT,   HID, 0, stream>>>(W1, A);
    fftmlp_main   <<<NN / 128, 256, 0, stream>>>(x1, x2, A, b1, v, cpt, out);
}
